// PFNetDense_19207093748411
// MI455X (gfx1250) — matrix operand order repaired and checked by reading, not compiled, not run on hardware
//
#include <hip/hip_runtime.h>
#include <hip/hip_bf16.h>

// ---------------------------------------------------------------------------
// PFNetDense (LSH-binned graph conv) for MI455X / gfx1250.
// BF16 WMMA (v_wmma_f32_16x16x32_bf16) for all GEMMs, f32 accumulation.
// Fragment-packed weights, transposed H, hoisted A-fragments.
// ---------------------------------------------------------------------------

#define BB   8        // batch
#define NN   12800    // nodes
#define DM   128      // msg feature dim
#define DN   256      // node feature dim
#define NBNS 100      // bins
#define BS   128      // bin size
#define NB2  199      // max bin index after mask offset (+99) plus 1

typedef __attribute__((ext_vector_type(16))) __bf16 v16bf;
typedef __attribute__((ext_vector_type(8)))  float  v8f;

__device__ __forceinline__ unsigned short f32_to_bf16(float f) {
    unsigned int u = __builtin_bit_cast(unsigned int, f);
    u += 0x7FFFu + ((u >> 16) & 1u);           // round-to-nearest-even
    return (unsigned short)(u >> 16);
}
__device__ __forceinline__ float bf16_to_f32(unsigned short h) {
    return __builtin_bit_cast(float, ((unsigned int)h) << 16);
}

__device__ __forceinline__ v16bf pack2(uint4 a, uint4 b) {
    union { uint4 q[2]; v16bf v; } u;
    u.q[0] = a; u.q[1] = b;
    return u.v;
}

// ---- CDNA5 wave32 WMMA fragment loaders (contiguous k order, same in A and B) --------------
// A (16x32 bf16): lane L holds M = L%16; K = 16*(L>=16) + e
//   -> 16 contiguous elements: two 16B loads
__device__ __forceinline__ v16bf ldA(const unsigned short* X, int ld,
                                     int m0, int k0, int lane) {
    const unsigned short* row =
        X + (m0 + (lane & 15)) * ld + k0 + ((lane & 16) ? 16 : 0);
    return pack2(*(const uint4*)row, *(const uint4*)(row + 8));
}
// B = X^T (X row-major M x K): lane holds N = L%16; K = 16*(L>=16) + e
//   -> 16 contiguous elements: two 16B loads
__device__ __forceinline__ v16bf ldBT(const unsigned short* X, int ld,
                                      int n0, int k0, int lane) {
    const unsigned short* row =
        X + (n0 + (lane & 15)) * ld + k0 + ((lane & 16) ? 16 : 0);
    return pack2(*(const uint4*)row, *(const uint4*)(row + 8));
}
// B fragment from fragment-packed weights: tile (kk,nt): 32 lanes x 16 elems
__device__ __forceinline__ v16bf ldBpk(const unsigned short* Wp,
                                       int kk, int nt, int lane) {
    const unsigned short* p = Wp + (((kk * 16 + nt) * 32) + lane) * 16;
    return pack2(*(const uint4*)p, *(const uint4*)(p + 8));
}
__device__ __forceinline__ v8f wmma_bf16(v16bf a, v16bf b, v8f c) {
    return __builtin_amdgcn_wmma_f32_16x16x32_bf16(
        false, a, false, b, (short)0, c, false, false);
}

// ---------------------------------------------------------------------------
// Kernel 1: weights -> bf16 fragment-packed layout: [theta | W_h | W_t]
//   wpk[w][tile=kk*16+nt][lane][e] with k = kk*32 + 16*(lane>=16) + e,
//                                       n = nt*16 + (lane&15)
// ---------------------------------------------------------------------------
__global__ void prep_weights(const float* __restrict__ theta,
                             const float* __restrict__ W_h,
                             const float* __restrict__ W_t,
                             unsigned short* __restrict__ wpk) {
    int idx = blockIdx.x * 256 + threadIdx.x;
    if (idx >= 3 * DN * DN) return;
    int which = idx >> 16;
    int rem   = idx & 0xFFFF;
    int tile  = rem >> 9;            // kk*16 + nt
    int lane  = (rem >> 4) & 31;
    int e     = rem & 15;
    int kk = tile >> 4, nt = tile & 15;
    int kcol = kk * 32 + ((lane & 16) ? 16 : 0) + e;
    int nrow = nt * 16 + (lane & 15);
    const float* src = (which == 0) ? theta : (which == 1) ? W_h : W_t;
    wpk[idx] = f32_to_bf16(src[kcol * DN + nrow]);
}

// ---------------------------------------------------------------------------
// Kernel 2: LSH bin assignment.  bin = argmax([X@C, -X@C]) (+99 if masked)
// ---------------------------------------------------------------------------
__global__ void bin_assign(const float* __restrict__ x_msg,
                           const float* __restrict__ codebook,
                           const unsigned char* __restrict__ msk,
                           int* __restrict__ bin_idx) {
    const int b     = blockIdx.x / 50;
    const int chunk = blockIdx.x % 50;
    const int tid   = threadIdx.x;
    __shared__ float cb[DM * 50];
    for (int i = tid; i < DM * 50; i += 256) {
        int d = i / 50, j = i % 50;
        cb[i] = codebook[d * NBNS + j];     // codebook is (128, 100) row-major
    }
    __syncthreads();

    const int n = chunk * 256 + tid;
    const float* xr = x_msg + ((size_t)b * NN + n) * DM;
    float acc[50];
#pragma unroll
    for (int j = 0; j < 50; ++j) acc[j] = 0.0f;
    for (int d = 0; d < DM; ++d) {
        float xv = xr[d];
        const float* c = &cb[d * 50];
#pragma unroll
        for (int j = 0; j < 50; ++j) acc[j] = fmaf(xv, c[j], acc[j]);
    }
    float best = acc[0]; int bi = 0;
#pragma unroll
    for (int j = 1; j < 50; ++j)
        if (acc[j] > best) { best = acc[j]; bi = j; }
#pragma unroll
    for (int j = 0; j < 50; ++j)
        if (-acc[j] > best) { best = -acc[j]; bi = 50 + j; }
    if (!msk[(size_t)b * NN + n]) bi += NBNS - 1;
    bin_idx[(size_t)b * NN + n] = bi;
}

// ---------------------------------------------------------------------------
// Kernel 3: stable counting sort -> perm[b][slot] = original node index
// ---------------------------------------------------------------------------
__global__ void bin_scatter(const int* __restrict__ bin_idx,
                            int* __restrict__ perm) {
    const int b = blockIdx.x, tid = threadIdx.x;
    __shared__ int base[NB2];
    __shared__ int cnt[NB2];
    __shared__ int cbins[256];
    for (int i = tid; i < NB2; i += 256) { base[i] = 0; cnt[i] = 0; }
    __syncthreads();
    for (int n = tid; n < NN; n += 256)
        atomicAdd(&base[bin_idx[(size_t)b * NN + n]], 1);
    __syncthreads();
    if (tid == 0) {
        int acc = 0;
        for (int i = 0; i < NB2; ++i) { int c = base[i]; base[i] = acc; acc += c; }
    }
    __syncthreads();
    for (int c = 0; c < NN / 256; ++c) {
        int n   = c * 256 + tid;
        int bin = bin_idx[(size_t)b * NN + n];
        cbins[tid] = bin;
        __syncthreads();
        int rank = 0;
        for (int i = 0; i < tid; ++i) rank += (cbins[i] == bin);
        atomicAdd(&cnt[bin], 1);
        __syncthreads();
        perm[(size_t)b * NN + base[bin] + rank] = n;
        __syncthreads();
        for (int i = tid; i < NB2; i += 256) { base[i] += cnt[i]; cnt[i] = 0; }
        __syncthreads();
    }
}

// ---------------------------------------------------------------------------
// Kernel 4: fused per-bin pipeline (800 blocks x 256 threads = 8 waves)
// Dynamic LDS: Xn(64K) | adj(32K) | HsT(64K, first half aliases Xm) | f32 tail
// ---------------------------------------------------------------------------
__global__ void fused_bin(const float* __restrict__ x_msg,
                          const float* __restrict__ x_node,
                          const float* __restrict__ b_t,
                          const unsigned char* __restrict__ msk,
                          const int* __restrict__ perm,
                          const unsigned short* __restrict__ wpk,
                          float* __restrict__ out) {
    const int b    = blockIdx.x / NBNS;
    const int k    = blockIdx.x % NBNS;
    const int tid  = threadIdx.x;
    const int lane = tid & 31;
    const int wv   = tid >> 5;           // wave id 0..7 (wave32)
    const int m0   = wv * 16;            // this wave's M tile

    extern __shared__ unsigned char smem_raw[];
    unsigned short* Xn  = (unsigned short*)smem_raw;   // 128x256 bf16
    unsigned short* adj = Xn + BS * DN;                // 128x128 bf16
    unsigned short* HsT = adj + BS * BS;               // 256x128 bf16 (transposed)
    unsigned short* Xm  = HsT;                         // alias: dead before HsT
    float* fb     = (float*)(HsT + DN * BS);
    float* na     = fb;                                // 128
    float* nrm    = fb + 128;                          // 128
    float* degs   = fb + 256;                          // 128
    float* mskrow = fb + 384;                          // 128
    int*   rows   = (int*)(fb + 512);                  // 128

    const unsigned short* thetapk = wpk;
    const unsigned short* whpk    = wpk + DN * DN;
    const unsigned short* wtpk    = wpk + 2 * DN * DN;

    // warm L2 with the packed weights (re-read by all 800 blocks)
    __builtin_prefetch(wpk + (size_t)tid * (3 * DN * DN / 256), 0, 1);

    // --- gather metadata -------------------------------------------------
    if (tid < BS) {
        int node = perm[(size_t)b * NN + k * BS + tid];
        rows[tid]   = node;
        mskrow[tid] = msk[(size_t)b * NN + node] ? 1.0f : 0.0f;
        degs[tid]   = 0.0f;
    }
    __syncthreads();

    // --- gather Xm (masked) and Xn (masked) as bf16 ----------------------
    for (int e = tid; e < BS * DM; e += 256) {
        int i = e >> 7, c = e & (DM - 1);
        float v = x_msg[((size_t)b * NN + rows[i]) * DM + c] * mskrow[i];
        Xm[i * DM + c] = f32_to_bf16(v);
    }
    for (int e = tid; e < BS * DN; e += 256) {
        int i = e >> 8, c = e & (DN - 1);
        float v = x_node[((size_t)b * NN + rows[i]) * DN + c] * mskrow[i];
        Xn[i * DN + c] = f32_to_bf16(v);
    }
    __syncthreads();

    // --- row squared norms (consistent with bf16 operands) ---------------
    if (tid < BS) {
        float s = 0.0f;
        for (int c = 0; c < DM; ++c) {
            float v = bf16_to_f32(Xm[tid * DM + c]);
            s = fmaf(v, v, s);
        }
        na[tid] = s;
    }
    __syncthreads();

    // --- adjacency: G = Xm Xm^T, adj = clip(exp(-0.1*sqrt(d2)))*mask -----
    {
        v16bf XmA[4];
#pragma unroll
        for (int kk = 0; kk < 4; ++kk) XmA[kk] = ldA(Xm, DM, m0, kk * 32, lane);
        for (int nt = 0; nt < 8; ++nt) {
            v8f acc = {};
#pragma unroll
            for (int kk = 0; kk < 4; ++kk)
                acc = wmma_bf16(XmA[kk], ldBT(Xm, DM, nt * 16, kk * 32, lane), acc);
            const int n  = nt * 16 + (lane & 15);
            const int mb = m0 + ((lane & 16) ? 8 : 0);
            const float nan_ = na[n];
            const float mskn = mskrow[n];
#pragma unroll
            for (int r = 0; r < 8; ++r) {
                int m = mb + r;
                float d2   = na[m] + nan_ - 2.0f * acc[r];
                float dist = __builtin_amdgcn_sqrtf(fmaxf(d2, 1e-6f));
                float av   = __expf(-0.1f * dist);
                av = fminf(fmaxf(av, 0.0f), 1.0f);
                av *= mskrow[m] * mskn;
                adj[m * BS + n] = f32_to_bf16(av);
                atomicAdd(&degs[m], av);               // ds_add_f32
            }
        }
    }
    __syncthreads();
    if (tid < BS) {
        float d = fminf(degs[tid], 1000.0f);
        nrm[tid] = rsqrtf(d + 1e-6f);
    }
    __syncthreads();

    // --- hoisted A fragments of Xn (shared by theta / W_h / W_t GEMMs) ----
    v16bf XnA[8];
#pragma unroll
    for (int kk = 0; kk < 8; ++kk) XnA[kk] = ldA(Xn, DN, m0, kk * 32, lane);

    // --- HsT[n][m] = ((Xn @ theta) * norm_row)^T  (overwrites Xm; dead) ---
    for (int nt = 0; nt < 16; ++nt) {
        v8f acc = {};
#pragma unroll
        for (int kk = 0; kk < 8; ++kk)
            acc = wmma_bf16(XnA[kk], ldBpk(thetapk, kk, nt, lane), acc);
        const int n  = nt * 16 + (lane & 15);
        const int mb = m0 + ((lane & 16) ? 8 : 0);
#pragma unroll
        for (int r = 0; r < 8; ++r) {
            int m = mb + r;
            HsT[n * BS + m] = f32_to_bf16(acc[r] * nrm[m]);
        }
    }
    __syncthreads();

    // --- F = adj @ Hs ; het = Xn @ W_h ; gate = sigmoid(Xn @ W_t + b_t) ---
    {
        v16bf adjA[4];
#pragma unroll
        for (int kk = 0; kk < 4; ++kk) adjA[kk] = ldA(adj, BS, m0, kk * 32, lane);
        for (int nt = 0; nt < 16; ++nt) {
            v8f aF = {}, aH = {}, aG = {};
#pragma unroll
            for (int kk = 0; kk < 4; ++kk)
                aF = wmma_bf16(adjA[kk], ldBT(HsT, BS, nt * 16, kk * 32, lane), aF);
#pragma unroll
            for (int kk = 0; kk < 8; ++kk) {
                aH = wmma_bf16(XnA[kk], ldBpk(whpk, kk, nt, lane), aH);
                aG = wmma_bf16(XnA[kk], ldBpk(wtpk, kk, nt, lane), aG);
            }
            const int n  = nt * 16 + (lane & 15);
            const int mb = m0 + ((lane & 16) ? 8 : 0);
            const float btn = b_t[n];
#pragma unroll
            for (int r = 0; r < 8; ++r) {
                int m = mb + r;
                float fhom = aF[r] * nrm[m];
                float gate = 1.0f / (1.0f + __expf(-(aG[r] + btn)));
                float v    = gate * fhom + (1.0f - gate) * aH[r];
                float o    = (v > 0.0f) ? v : (__expf(v) - 1.0f);
                out[((size_t)b * NN + rows[m]) * DN + n] = o * mskrow[m];
            }
        }
    }
}

// ---------------------------------------------------------------------------
extern "C" void kernel_launch(void* const* d_in, const int* in_sizes, int n_in,
                              void* d_out, int out_size, void* d_ws, size_t ws_size,
                              hipStream_t stream) {
    const float*         x_msg    = (const float*)d_in[0];
    const float*         x_node   = (const float*)d_in[1];
    const float*         codebook = (const float*)d_in[2];
    const float*         W_t      = (const float*)d_in[3];
    const float*         b_t      = (const float*)d_in[4];
    const float*         W_h      = (const float*)d_in[5];
    const float*         theta    = (const float*)d_in[6];
    const unsigned char* msk      = (const unsigned char*)d_in[7];
    float*               out      = (float*)d_out;

    char* ws = (char*)d_ws;
    int*            perm    = (int*)ws;                         // B*N ints
    int*            bin_idx = (int*)(ws + (size_t)BB * NN * 4); // B*N ints
    unsigned short* wpk     = (unsigned short*)(ws + (size_t)2 * BB * NN * 4);

    prep_weights<<<(3 * DN * DN + 255) / 256, 256, 0, stream>>>(theta, W_h, W_t, wpk);
    bin_assign  <<<BB * (NN / 256), 256, 0, stream>>>(x_msg, codebook, msk, bin_idx);
    bin_scatter <<<BB, 256, 0, stream>>>(bin_idx, perm);

    // Xn (64K) + adj (32K) + HsT (64K) + 512 f32 + 128 int = 166400 bytes
    const size_t lds = (size_t)(BS * DN + BS * BS + DN * BS) * 2 + 512 * 4 + 128 * 4;
    (void)hipFuncSetAttribute((const void*)fused_bin,
                              hipFuncAttributeMaxDynamicSharedMemorySize, (int)lds);
    fused_bin<<<BB * NBNS, 256, lds, stream>>>(x_msg, x_node, b_t, msk, perm, wpk, out);
}
